// MoEFFN_2808908611853
// MI455X (gfx1250) — matrix operand order repaired and checked by reading, not compiled, not run on hardware
//
#include <hip/hip_runtime.h>
#include <hip/hip_bf16.h>
#include <math.h>

// ---------------------------------------------------------------------------
// MoE FFN for MI455X (gfx1250): top-2 routed experts, bf16 WMMA w/ f32 accum.
// ---------------------------------------------------------------------------

constexpr int kB  = 8;
constexpr int kS  = 1024;
constexpr int kN  = kB * kS;   // 8192 tokens
constexpr int kH  = 1024;
constexpr int kFF = 4096;
constexpr int kE  = 8;
constexpr int TM  = 16;        // tokens per block tile
constexpr int FCH = 128;       // FF chunk per iteration

typedef __attribute__((ext_vector_type(16))) __bf16 v16bf;
typedef __attribute__((ext_vector_type(8)))  __bf16 v8bf;
typedef __attribute__((ext_vector_type(8)))  float  v8f;
typedef __attribute__((ext_vector_type(2)))  unsigned long long ull2;

__device__ __forceinline__ unsigned short f2bf(float f) {
    unsigned u = __float_as_uint(f);
    u += 0x7FFFu + ((u >> 16) & 1u);      // round-to-nearest-even
    return (unsigned short)(u >> 16);
}

__device__ __forceinline__ v8f vzero8() {
    v8f z;
    #pragma unroll
    for (int i = 0; i < 8; ++i) z[i] = 0.0f;
    return z;
}

// Build a 16-element bf16 A/B fragment from two 16-byte chunks.
__device__ __forceinline__ v16bf ld_frag(const unsigned short* p0,
                                         const unsigned short* p1) {
    v8bf lo = *(const v8bf*)p0;
    v8bf hi = *(const v8bf*)p1;
    return __builtin_shufflevector(lo, hi, 0,1,2,3,4,5,6,7,8,9,10,11,12,13,14,15);
}

// Direct memory -> LDS copy, 16B, tracked by ASYNCcnt (no VGPR round-trip).
// Low 32 bits of a generic pointer to LDS are the LDS byte offset (ISA 10.2).
__device__ __forceinline__ void async_g2l_b128(void* lds_ptr, const void* gptr) {
    unsigned ldsa = (unsigned)(unsigned long long)(uintptr_t)lds_ptr;
    unsigned long long ga = (unsigned long long)(uintptr_t)gptr;
    asm volatile("global_load_async_to_lds_b128 %0, %1, off"
                 :: "v"(ldsa), "v"(ga) : "memory");
}
__device__ __forceinline__ void wait_asynccnt0() {
    asm volatile("s_wait_asynccnt 0x0" ::: "memory");
}

// ---------------------------------------------------------------------------
// 0) zero the per-expert counters
// ---------------------------------------------------------------------------
__global__ void init_kernel(int* cnt) {
    if (threadIdx.x < kE) cnt[threadIdx.x] = 0;
}

// ---------------------------------------------------------------------------
// 1) f32 -> bf16 bulk conversion (grid-stride)
// ---------------------------------------------------------------------------
__global__ void convert_kernel(const float* __restrict__ src,
                               unsigned short* __restrict__ dst, size_t n) {
    size_t i = (size_t)blockIdx.x * blockDim.x + threadIdx.x;
    size_t stride = (size_t)gridDim.x * blockDim.x;
    for (; i < n; i += stride) dst[i] = f2bf(src[i]);
}

// ---------------------------------------------------------------------------
// 2) router: logits, top-2 softmax, build per-expert (token,slot,weight) lists
//    one wave32 per token
// ---------------------------------------------------------------------------
__global__ __launch_bounds__(256) void router_kernel(
        const float* __restrict__ x, const float* __restrict__ rw,
        const float* __restrict__ rb, int* __restrict__ cnt,
        int* __restrict__ idxL, float* __restrict__ wL) {
    const int lane = threadIdx.x & 31;
    const int wave = threadIdx.x >> 5;
    const int tok  = blockIdx.x * 8 + wave;
    if (tok >= kN) return;
    const float* xp = x + (size_t)tok * kH;

    float acc[kE];
    #pragma unroll
    for (int e = 0; e < kE; ++e) acc[e] = 0.0f;
    for (int h = lane; h < kH; h += 32) {
        float xv = xp[h];
        #pragma unroll
        for (int e = 0; e < kE; ++e) acc[e] += xv * rw[e * kH + h];
    }
    #pragma unroll
    for (int e = 0; e < kE; ++e)
        for (int off = 16; off > 0; off >>= 1)
            acc[e] += __shfl_xor(acc[e], off, 32);

    if (lane == 0) {
        float l[kE];
        #pragma unroll
        for (int e = 0; e < kE; ++e) l[e] = acc[e] + rb[e];
        int bi = 0; float best = l[0];
        #pragma unroll
        for (int e = 1; e < kE; ++e) if (l[e] > best) { best = l[e]; bi = e; }
        int si = -1; float sec = -1e30f;
        #pragma unroll
        for (int e = 0; e < kE; ++e)
            if (e != bi && l[e] > sec) { sec = l[e]; si = e; }
        float p2 = expf(sec - best);
        float d  = 1.0f + p2;
        float wA = 1.0f / d, wB = p2 / d;
        int p;
        p = atomicAdd(&cnt[bi], 1);
        idxL[bi * kN + p] = (tok << 1) | 0; wL[bi * kN + p] = wA;
        p = atomicAdd(&cnt[si], 1);
        idxL[si * kN + p] = (tok << 1) | 1; wL[si * kN + p] = wB;
    }
}

// ---------------------------------------------------------------------------
// 3) fused expert FFN:  y = gelu(X w1^T + b1) w2^T + b2, scaled by combine wt
//    block = (tile of 16 tokens) x expert; 8 waves; bf16 WMMA, f32 accum
// ---------------------------------------------------------------------------
__global__ __launch_bounds__(256) void ffn_kernel(
        const unsigned short* __restrict__ Xbf,
        const unsigned short* __restrict__ w1bf,
        const unsigned short* __restrict__ w2bf,
        const float* __restrict__ b1, const float* __restrict__ b2,
        const int* __restrict__ cnt, const int* __restrict__ idxL,
        const float* __restrict__ wL, float* __restrict__ ybuf) {
    const int e    = blockIdx.y;
    const int tile = blockIdx.x;
    const int cntE = cnt[e];
    if (tile * TM >= cntE) return;

    __shared__ unsigned short Xs[TM][kH + 8];     // X tile (bf16), padded
    __shared__ unsigned short H1s[TM][FCH + 8];   // gelu intermediate (bf16)
    __shared__ int   sTok[TM];
    __shared__ float sW[TM];

    const int tid  = threadIdx.x;
    const int lane = tid & 31;
    const int wave = tid >> 5;
    const int m16  = lane & 15;
    const int kh   = lane >> 4;     // which K-half this lane holds

    if (tid < TM) {
        int p = tile * TM + tid;
        if (p < cntE) { sTok[tid] = idxL[e * kN + p]; sW[tid] = wL[e * kN + p]; }
        else          { sTok[tid] = -1;               sW[tid] = 0.0f; }
    }
    __syncthreads();

    // gather the 16-token X tile straight into LDS via async b128 copies
    for (int i = tid; i < TM * (kH / 8); i += 256) {
        int m  = i >> 7;            // kH/8 = 128 chunks per row
        int c8 = (i & 127) * 8;
        int pk = sTok[m];
        if (pk >= 0) {
            async_g2l_b128(&Xs[m][c8], Xbf + (size_t)(pk >> 1) * kH + c8);
        } else {
            ull2 z; z[0] = 0; z[1] = 0;
            *(ull2*)(&Xs[m][c8]) = z;   // padded slot: zero-fill (DS path)
        }
    }
    wait_asynccnt0();
    __syncthreads();

    const size_t w1base = (size_t)e * kFF * kH;
    const size_t w2base = (size_t)e * kH * kFF;

    v8f acc[8];                     // 16x128 f32 output slice per wave
    #pragma unroll
    for (int j = 0; j < 8; ++j) acc[j] = vzero8();

    for (int ffBase = 0; ffBase < kFF; ffBase += FCH) {
        // ---- phase A: H1 = gelu(X @ w1^T + b1), one 16x16 tile per wave ----
        // two independent accumulators to break the WMMA RAW chain
        {
            v8f c0 = vzero8(), c1 = vzero8();
            const int ffcol = ffBase + wave * 16 + m16;   // B column (n) = C column
            const unsigned short* wp = w1bf + w1base + (size_t)ffcol * kH + kh * 16;
            for (int k0 = 0; k0 < kH; k0 += 64) {
                __builtin_prefetch(wp + k0 + 512, 0, 1);  // global_prefetch_b8
                v16bf a0 = ld_frag(&Xs[m16][k0 + kh * 16],
                                   &Xs[m16][k0 + 8 + kh * 16]);
                v16bf b0 = *(const v16bf*)(wp + k0);
                c0 = __builtin_amdgcn_wmma_f32_16x16x32_bf16(
                         false, a0, false, b0, (short)0, c0, false, false);
                v16bf a1 = ld_frag(&Xs[m16][k0 + 32 + kh * 16],
                                   &Xs[m16][k0 + 40 + kh * 16]);
                v16bf b1v = *(const v16bf*)(wp + k0 + 32);
                c1 = __builtin_amdgcn_wmma_f32_16x16x32_bf16(
                         false, a1, false, b1v, (short)0, c1, false, false);
            }
            const float bias = b1[e * kFF + ffcol];
            const int mrow = kh * 8;
            const int ccol = wave * 16 + m16;
            #pragma unroll
            for (int r = 0; r < 8; ++r) {
                float g = c0[r] + c1[r] + bias;
                g = 0.5f * g * (1.0f + erff(g * 0.70710678118654752f));
                H1s[mrow + r][ccol] = f2bf(g);
            }
        }
        __syncthreads();
        // ---- phase B: acc += H1 @ w2^T, wave owns H columns [wave*128, +128) ----
        #pragma unroll
        for (int kk = 0; kk < FCH; kk += 32) {
            v16bf a = ld_frag(&H1s[m16][kk + kh * 16], &H1s[m16][kk + 8 + kh * 16]);
            #pragma unroll
            for (int j = 0; j < 8; ++j) {
                const int hcol = wave * 128 + j * 16 + m16;
                const unsigned short* wp2 =
                    w2bf + w2base + (size_t)hcol * kFF + ffBase + kk + kh * 16;
                v16bf b = *(const v16bf*)wp2;
                acc[j] = __builtin_amdgcn_wmma_f32_16x16x32_bf16(
                             false, a, false, b, (short)0, acc[j], false, false);
            }
        }
        __syncthreads();
    }

    // ---- epilogue: (acc + b2) * combine_weight -> exclusive slot buffer ----
    const int mrow = kh * 8;
    #pragma unroll
    for (int j = 0; j < 8; ++j) {
        const int hcol = wave * 128 + j * 16 + m16;
        const float bias2 = b2[e * kH + hcol];
        #pragma unroll
        for (int r = 0; r < 8; ++r) {
            int m  = mrow + r;
            int pk = sTok[m];
            if (pk >= 0)
                ybuf[(size_t)pk * kH + hcol] = (acc[j][r] + bias2) * sW[m];
        }
    }
}

// ---------------------------------------------------------------------------
// 4) residual + combine slots + LayerNorm -> out
// ---------------------------------------------------------------------------
__global__ __launch_bounds__(256) void ln_kernel(
        const float* __restrict__ x, const float* __restrict__ ybuf,
        const float* __restrict__ lw, const float* __restrict__ lb,
        float* __restrict__ out) {
    const int tok = blockIdx.x;
    const int tid = threadIdx.x;
    __shared__ float red[8];
    const float* xp = x    + (size_t)tok * kH;
    const float* y0 = ybuf + (size_t)(tok * 2) * kH;
    const float* y1 = y0 + kH;

    float v[4]; float s = 0.0f;
    #pragma unroll
    for (int i = 0; i < 4; ++i) {
        int h = tid + i * 256;
        float t = xp[h] + y0[h] + y1[h];
        v[i] = t; s += t;
    }
    for (int off = 16; off > 0; off >>= 1) s += __shfl_xor(s, off, 32);
    if ((tid & 31) == 0) red[tid >> 5] = s;
    __syncthreads();
    float tot = 0.0f;
    #pragma unroll
    for (int i = 0; i < 8; ++i) tot += red[i];
    const float mean = tot * (1.0f / kH);
    __syncthreads();

    float s2 = 0.0f;
    #pragma unroll
    for (int i = 0; i < 4; ++i) { float d = v[i] - mean; s2 += d * d; }
    for (int off = 16; off > 0; off >>= 1) s2 += __shfl_xor(s2, off, 32);
    if ((tid & 31) == 0) red[tid >> 5] = s2;
    __syncthreads();
    float tot2 = 0.0f;
    #pragma unroll
    for (int i = 0; i < 8; ++i) tot2 += red[i];
    const float rs = rsqrtf(tot2 * (1.0f / kH) + 1e-12f);

    #pragma unroll
    for (int i = 0; i < 4; ++i) {
        int h = tid + i * 256;
        out[(size_t)tok * kH + h] = (v[i] - mean) * rs * lw[h] + lb[h];
    }
}

// ---------------------------------------------------------------------------
extern "C" void kernel_launch(void* const* d_in, const int* in_sizes, int n_in,
                              void* d_out, int out_size, void* d_ws, size_t ws_size,
                              hipStream_t stream) {
    const float* hidden   = (const float*)d_in[0];
    const float* router_w = (const float*)d_in[1];
    const float* router_b = (const float*)d_in[2];
    const float* w1       = (const float*)d_in[3];
    const float* b1       = (const float*)d_in[4];
    const float* w2       = (const float*)d_in[5];
    const float* b2       = (const float*)d_in[6];
    const float* ln_w     = (const float*)d_in[7];
    const float* ln_b     = (const float*)d_in[8];
    float* out = (float*)d_out;

    char* ws = (char*)d_ws;
    size_t off = 0;
    auto alloc = [&](size_t bytes) -> void* {
        off = (off + 255) & ~(size_t)255;
        void* p = ws + off;
        off += bytes;
        return p;
    };
    unsigned short* Xbf  = (unsigned short*)alloc((size_t)kN * kH * 2);
    unsigned short* w1bf = (unsigned short*)alloc((size_t)kE * kFF * kH * 2);
    unsigned short* w2bf = (unsigned short*)alloc((size_t)kE * kH * kFF * 2);
    float*          ybuf = (float*)         alloc((size_t)kN * 2 * kH * 4);
    int*            cnt  = (int*)           alloc(256);
    int*            idxL = (int*)           alloc((size_t)kE * kN * 4);
    float*          wL   = (float*)         alloc((size_t)kE * kN * 4);

    init_kernel<<<1, 32, 0, stream>>>(cnt);
    convert_kernel<<<2048, 256, 0, stream>>>(hidden, Xbf, (size_t)kN * kH);
    convert_kernel<<<4096, 256, 0, stream>>>(w1, w1bf, (size_t)kE * kFF * kH);
    convert_kernel<<<4096, 256, 0, stream>>>(w2, w2bf, (size_t)kE * kH * kFF);
    router_kernel<<<kN / 8, 256, 0, stream>>>(hidden, router_w, router_b,
                                              cnt, idxL, wL);
    dim3 g(kN / TM, kE);
    ffn_kernel<<<g, 256, 0, stream>>>(Xbf, w1bf, w2bf, b1, b2,
                                      cnt, idxL, wL, ybuf);
    ln_kernel<<<kN, 256, 0, stream>>>(hidden, ybuf, ln_w, ln_b, out);
}
